// HMC_80418967650382
// MI455X (gfx1250) — matrix operand order repaired and checked by reading, not compiled, not run on hardware
//
#include <hip/hip_runtime.h>

typedef __attribute__((ext_vector_type(16))) __bf16 v16bf;
typedef __attribute__((ext_vector_type(2)))  __bf16 v2bf;
typedef __attribute__((ext_vector_type(8)))  float  v8f;

#define THREADS 256
#define BK 32
#define LDT (BK + 8)   // LDS row stride in bf16 units (80 bytes): 16B aligned, conflict-free

// f32 -> bf16 via native hardware convert (lowers to v_cvt_pk_bf16_f32 / v_cvt_bf16_f32)
__device__ __forceinline__ unsigned pack2bf(float a, float b) {
#if __has_builtin(__builtin_amdgcn_cvt_pk_bf16_f32)
  v2bf p = __builtin_amdgcn_cvt_pk_bf16_f32(a, b);
  return __builtin_bit_cast(unsigned, p);
#else
  v2bf p;
  p[0] = (__bf16)a;
  p[1] = (__bf16)b;
  return __builtin_bit_cast(unsigned, p);
#endif
}
__device__ __forceinline__ unsigned short f2bf(float f) {
  __bf16 h = (__bf16)f;
  return __builtin_bit_cast(unsigned short, h);
}

// A fragment (16x32 bf16): lane L holds row L&15; VGPR0-3 = K kb..kb+7, VGPR4-7 = K kb+16..kb+23, kb = (L>>4)*8
__device__ __forceinline__ v16bf load_frag_a(const unsigned short* row, int kb) {
  v16bf v;
  ((uint4*)&v)[0] = *(const uint4*)(row + kb);
  ((uint4*)&v)[1] = *(const uint4*)(row + kb + 16);
  return v;
}
// B fragment (32x16 bf16): lane L holds col L&15; VGPR0-3 = K kb..kb+7, VGPR4-7 = K kb+16..kb+23, kb = (L>>4)*8
__device__ __forceinline__ v16bf load_frag_b(const unsigned short* row, int kb) {
  v16bf v;
  ((uint4*)&v)[0] = *(const uint4*)(row + kb);
  ((uint4*)&v)[1] = *(const uint4*)(row + kb + 16);
  return v;
}

// C = relu(A @ W + bias); A is [M,K] (f32 or bf16, row stride lda),
// W is [K,N] f32 row-major (ldb), out row stride ldo, bf16 or f32 output.
template<int BM, int BN, int WM, int WN, bool A_F32, bool OUT_F32>
__global__ __launch_bounds__(THREADS)
void gemm_wmma(const void* __restrict__ Av, int lda,
               const float* __restrict__ Bw, int ldb,
               const float* __restrict__ bias,
               void* __restrict__ Ov, int ldo, int K) {
  constexpr int TM  = WM / 16, TN = WN / 16;
  constexpr int NWN = BN / WN;
  static_assert((BM / WM) * NWN == THREADS / 32, "8 waves required");
  constexpr int ITERS_A = (BM * BK) / (THREADS * 4);      // 4-element loads per thread
  constexpr int ROWS_A  = THREADS / (BK / 4);             // 32 rows per pass
  constexpr int ITERS_B = (BN * BK) / (THREADS * 4);
  constexpr int KSPAN_B = (THREADS * 4) / BN;             // k covered per B pass

  __shared__ unsigned short Asl[2][BM * LDT];
  __shared__ unsigned short Bsl[2][BN * LDT];

  const int t    = threadIdx.x;
  const int lane = t & 31;
  const int wave = t >> 5;
  const int wm   = (wave / NWN) * WM;
  const int wn   = (wave % NWN) * WN;
  const int m0   = blockIdx.y * BM;
  const int n0   = blockIdx.x * BN;

  const float*          Af = (const float*)Av;
  const unsigned short* Ah = (const unsigned short*)Av;

  float4 raf[ITERS_A];
  uint2  rah[ITERS_A];
  float  rb[ITERS_B][4];

  auto gload = [&](int k0) {
    const int arow = t >> 3, acol = (t & 7) * 4;
#pragma unroll
    for (int i = 0; i < ITERS_A; ++i) {
      size_t base = (size_t)(m0 + arow + i * ROWS_A) * lda + k0 + acol;
      if constexpr (A_F32) raf[i] = *(const float4*)(Af + base);
      else                 rah[i] = *(const uint2*)(Ah + base);
    }
    const int bn = t & (BN - 1);
#pragma unroll
    for (int i = 0; i < ITERS_B; ++i) {
      int kq = (t / BN) * 4 + i * KSPAN_B;
#pragma unroll
      for (int j = 0; j < 4; ++j)
        rb[i][j] = Bw[(size_t)(k0 + kq + j) * ldb + n0 + bn];
    }
  };

  auto sstore = [&](int buf) {
    const int arow = t >> 3, acol = (t & 7) * 4;
#pragma unroll
    for (int i = 0; i < ITERS_A; ++i) {
      uint2 p;
      if constexpr (A_F32) { p.x = pack2bf(raf[i].x, raf[i].y); p.y = pack2bf(raf[i].z, raf[i].w); }
      else                 p = rah[i];
      *(uint2*)&Asl[buf][(arow + i * ROWS_A) * LDT + acol] = p;
    }
    const int bn = t & (BN - 1);
#pragma unroll
    for (int i = 0; i < ITERS_B; ++i) {
      int kq = (t / BN) * 4 + i * KSPAN_B;
      uint2 p;
      p.x = pack2bf(rb[i][0], rb[i][1]);
      p.y = pack2bf(rb[i][2], rb[i][3]);
      *(uint2*)&Bsl[buf][bn * LDT + kq] = p;
    }
  };

  v8f acc[TM][TN] = {};

  gload(0);
  sstore(0);
  __syncthreads();

  const int KT = K / BK;
  const int ka = (lane >> 4) << 3;
  const int kb = (lane >> 4) << 3;
  for (int kt = 0; kt < KT; ++kt) {
    const int cur = kt & 1;
    if (kt + 1 < KT) gload((kt + 1) * BK);   // overlap next-tile global loads with WMMA

    v16bf afr[TM], bfr[TN];
#pragma unroll
    for (int i = 0; i < TM; ++i)
      afr[i] = load_frag_a(&Asl[cur][(wm + i * 16 + (lane & 15)) * LDT], ka);
#pragma unroll
    for (int j = 0; j < TN; ++j)
      bfr[j] = load_frag_b(&Bsl[cur][(wn + j * 16 + (lane & 15)) * LDT], kb);
#pragma unroll
    for (int i = 0; i < TM; ++i)
#pragma unroll
      for (int j = 0; j < TN; ++j)
        acc[i][j] = __builtin_amdgcn_wmma_f32_16x16x32_bf16(
            false, afr[i], false, bfr[j], (short)0, acc[i][j], false, false);

    if (kt + 1 < KT) sstore((kt + 1) & 1);
    __syncthreads();
  }

  float*          Of = (float*)Ov;
  unsigned short* Oh = (unsigned short*)Ov;
#pragma unroll
  for (int i = 0; i < TM; ++i) {
#pragma unroll
    for (int j = 0; j < TN; ++j) {
      const int c  = n0 + wn + j * 16 + (lane & 15);
      const int r0 = m0 + wm + i * 16 + ((lane >> 4) << 3);
      const float bv = bias[c];
#pragma unroll
      for (int v = 0; v < 8; ++v) {
        float val = fmaxf(acc[i][j][v] + bv, 0.0f);
        size_t o = (size_t)(r0 + v) * ldo + c;
        if constexpr (OUT_F32) Of[o] = val;
        else                   Oh[o] = f2bf(val);
      }
    }
  }
}

// One wave per row: softmax(L1[64]) + argmax -> parent; mask L2[256] by child_parent; softmax(L2). In place.
__global__ __launch_bounds__(256)
void hier_softmax(float* __restrict__ L1, float* __restrict__ L2,
                  const int* __restrict__ child_parent) {
  const int row  = blockIdx.x * 8 + (threadIdx.x >> 5);
  const int lane = threadIdx.x & 31;

  float* l1 = L1 + (size_t)row * 64;
  const float a = l1[lane], b = l1[lane + 32];
  float m; int idx;
  if (b > a) { m = b; idx = lane + 32; } else { m = a; idx = lane; }
#pragma unroll
  for (int off = 16; off >= 1; off >>= 1) {
    float om = __shfl_xor(m, off, 32);
    int   oi = __shfl_xor(idx, off, 32);
    if (om > m || (om == m && oi < idx)) { m = om; idx = oi; }  // first-index ties like jnp.argmax
  }
  const float ea = expf(a - m), eb = expf(b - m);
  float s = ea + eb;
#pragma unroll
  for (int off = 16; off >= 1; off >>= 1) s += __shfl_xor(s, off, 32);
  const float inv = 1.0f / s;
  l1[lane]      = ea * inv;
  l1[lane + 32] = eb * inv;
  const int parent = idx;

  float* l2 = L2 + (size_t)row * 256;
  float v[8];
  float m2 = -3.4e38f;
#pragma unroll
  for (int j = 0; j < 8; ++j) {
    const int c = j * 32 + lane;
    float x = l2[c] + ((child_parent[c] == parent) ? 0.0f : -10000.0f);
    v[j] = x;
    m2 = fmaxf(m2, x);
  }
#pragma unroll
  for (int off = 16; off >= 1; off >>= 1) m2 = fmaxf(m2, __shfl_xor(m2, off, 32));
  float s2 = 0.0f;
#pragma unroll
  for (int j = 0; j < 8; ++j) { v[j] = expf(v[j] - m2); s2 += v[j]; }
#pragma unroll
  for (int off = 16; off >= 1; off >>= 1) s2 += __shfl_xor(s2, off, 32);
  const float inv2 = 1.0f / s2;
#pragma unroll
  for (int j = 0; j < 8; ++j) l2[j * 32 + lane] = v[j] * inv2;
}

extern "C" void kernel_launch(void* const* d_in, const int* in_sizes, int n_in,
                              void* d_out, int out_size, void* d_ws, size_t ws_size,
                              hipStream_t stream) {
  const float* x    = (const float*)d_in[0];
  const float* W1_1 = (const float*)d_in[1];
  const float* b1_1 = (const float*)d_in[2];
  const float* W1_2 = (const float*)d_in[3];
  const float* b1_2 = (const float*)d_in[4];
  const float* W2_1 = (const float*)d_in[5];
  const float* b2_1 = (const float*)d_in[6];
  const float* W2_2 = (const float*)d_in[7];
  const float* b2_2 = (const float*)d_in[8];
  const int* child_parent = (const int*)d_in[9];

  constexpr int B = 8192, F = 1024, H = 2048, L1N = 64, L2N = 256;

  unsigned short* hcat = (unsigned short*)d_ws;       // [B, 2H] bf16: h1 | h2
  float* L1o = (float*)d_out;                          // [B, 64]
  float* L2o = L1o + (size_t)B * L1N;                  // [B, 256]

  dim3 blk(THREADS);
  // h1 = relu(x @ W1_1 + b1_1) -> hcat[:, 0:2048]  (bf16)
  gemm_wmma<128, 128, 64, 32, true, false>
      <<<dim3(H / 128, B / 128), blk, 0, stream>>>(x, F, W1_1, H, b1_1, hcat, 2 * H, F);
  // h2 = relu(x @ W2_1 + b2_1) -> hcat[:, 2048:4096]
  gemm_wmma<128, 128, 64, 32, true, false>
      <<<dim3(H / 128, B / 128), blk, 0, stream>>>(x, F, W2_1, H, b2_1, hcat + H, 2 * H, F);
  // L1 logits = relu(h1 @ W1_2 + b1_2) -> d_out[:, :64]  (f32, K = 2048 over hcat left half)
  gemm_wmma<128, 64, 32, 32, false, true>
      <<<dim3(L1N / 64, B / 128), blk, 0, stream>>>(hcat, 2 * H, W1_2, L1N, b1_2, L1o, L1N, H);
  // L2 logits = relu(cat @ W2_2 + b2_2) -> d_out L2 region (f32, K = 4096 over full hcat)
  gemm_wmma<128, 128, 64, 32, false, true>
      <<<dim3(L2N / 128, B / 128), blk, 0, stream>>>(hcat, 2 * H, W2_2, L2N, b2_2, L2o, L2N, 2 * H);
  // fused hierarchical softmax epilogue (in place on d_out)
  hier_softmax<<<dim3(B / 8), blk, 0, stream>>>(L1o, L2o, child_parent);
}
